// MGNN_73821897883898
// MI455X (gfx1250) — matrix operand order repaired and checked by reading, not compiled, not run on hardware
//
#include <hip/hip_runtime.h>
#include <hip/hip_bf16.h>

#define DEV __device__ __forceinline__

typedef __bf16 bf16;
typedef bf16  bf16x8  __attribute__((ext_vector_type(8)));
typedef bf16  bf16x16 __attribute__((ext_vector_type(16)));
typedef float f32x4   __attribute__((ext_vector_type(4)));
typedef float f32x8   __attribute__((ext_vector_type(8)));

#define N_NODES 10000
#define N_EDGES 120000
#define N_TRIP  600000
#define FDIM    128
#define CBETA   8
#define NLAYERS 3
#define RCUT    5.0f

// silu via hardware reciprocal (v_rcp_f32, TRANS pipe) instead of IEEE divide
DEV float silu_f(float x){ return x * __builtin_amdgcn_rcpf(1.0f + __expf(-x)); }
DEV void wave_ds_fence(){ asm volatile("s_wait_dscnt 0" ::: "memory"); }

// A fragment (16x32 bf16, M x K) from LDS, row-major with leading dim ld.
// Per ISA: lane L -> M = L%16, h = L/16; elems 0..7 -> K = h*8+0..7 ; elems 8..15 -> K = 16+h*8+0..7
DEV bf16x16 a_frag(const bf16* A, int ld, int kbase, int lane){
  int m = lane & 15, h = lane >> 4;
  const bf16* p = A + m*ld + kbase + h*8;
  bf16x8 lo = *(const bf16x8*)(p);
  bf16x8 hi = *(const bf16x8*)(p + 16);
  bf16x16 r;
#pragma unroll
  for (int t = 0; t < 8; ++t){ r[t] = lo[t]; r[t+8] = hi[t]; }
  return r;
}

// B fragment (32x16 bf16, K x N) from global transposed weights WT[n*K + k].
// lane L -> N = L%16, h = L/16; elems 0..7 -> K = h*8+0..7 ; elems 8..15 -> K = 16+h*8+0..7
DEV bf16x16 b_frag(const bf16* WT, int K, int nbase, int kbase, int lane){
  int n = lane & 15, h = lane >> 4;
  const bf16* p = WT + (size_t)(nbase + n)*K + kbase + h*8;
  bf16x8 lo = *(const bf16x8*)(p);
  bf16x8 hi = *(const bf16x8*)(p + 16);
  bf16x16 r;
#pragma unroll
  for (int t = 0; t < 8; ++t){ r[t] = lo[t]; r[t+8] = hi[t]; }
  return r;
}

// 16x16 output tile: D = A(16xK, LDS bf16) * W(KxNout via transposed WT)
DEV f32x8 gemm_tile(const bf16* A, int ld, int K, const bf16* WT, int nbase, int lane){
  f32x8 acc;
#pragma unroll
  for (int r = 0; r < 8; ++r) acc[r] = 0.0f;
  for (int k = 0; k < K; k += 32){
    __builtin_prefetch(WT + (size_t)(nbase + (lane & 15))*K + k + 32, 0, 1);
    bf16x16 a = a_frag(A, ld, k, lane);
    bf16x16 b = b_frag(WT, K, nbase, k, lane);
    acc = __builtin_amdgcn_wmma_f32_16x16x32_bf16(false, a, false, b, (short)0, acc, false, false);
  }
  return acc;
}

// ---------------- prep kernels ----------------

__global__ void k_cvtT(const float* __restrict__ W, bf16* __restrict__ WT, int K, int Nout){
  int idx = blockIdx.x*blockDim.x + threadIdx.x;
  if (idx >= K*Nout) return;
  int k = idx / Nout, n = idx - k*Nout;
  WT[(size_t)n*K + k] = (bf16)W[idx];
}

__global__ void k_cheb(const float* __restrict__ dist, float* __restrict__ Q, int E){
  int e = blockIdx.x*blockDim.x + threadIdx.x;
  if (e >= E) return;
  float d = dist[e];
  float x = 2.0f*d/RCUT - 1.0f;
  float env = (d - RCUT)*(d - RCUT);
  float mask = (d < RCUT) ? 1.0f : 0.0f;
  float v[CBETA]; v[0] = 1.0f; v[1] = x;
#pragma unroll
  for (int b = 2; b < CBETA; ++b) v[b] = 2.0f*x*v[b-1] - v[b-2];
#pragma unroll
  for (int b = 0; b < CBETA; ++b) Q[(size_t)e*CBETA + b] = v[b]*env*mask;
}

__global__ void k_embed(const int* __restrict__ an, const float* __restrict__ emb,
                        float* __restrict__ z, int N){
  int idx = blockIdx.x*blockDim.x + threadIdx.x;
  if (idx >= N*FDIM) return;
  int n = idx / FDIM, f = idx - n*FDIM;
  z[idx] = emb[(size_t)an[n]*FDIM + f];
}

__global__ void k_zero(float* __restrict__ p, size_t n){
  size_t i = (size_t)blockIdx.x*blockDim.x + threadIdx.x;
  size_t st = (size_t)gridDim.x*blockDim.x;
  for (; i < n; i += st) p[i] = 0.0f;
}

// ---------------- per-layer kernels ----------------

// z1 = silu(z@W1+b1)@W2+b2  -> bf16 (only consumed by edge gather)
__global__ void __launch_bounds__(128) k_node_mlp(
    const float* __restrict__ z, const bf16* __restrict__ W1T, const float* __restrict__ b1,
    const bf16* __restrict__ W2T, const float* __restrict__ b2,
    bf16* __restrict__ z1b, int N)
{
  __shared__ __align__(32) bf16 sA[4][16*FDIM];
  __shared__ __align__(32) bf16 sH[4][16*FDIM];
  int wave = threadIdx.x >> 5, lane = threadIdx.x & 31;
  bf16* A = sA[wave]; bf16* H = sH[wave];
  int n0 = (blockIdx.x*4 + wave)*16;
  { // stage 16 rows of z as bf16 (2 lanes per row)
    int m = lane >> 1, p = lane & 1;
    int row = n0 + m; if (row > N-1) row = N-1;
    const float* s = z + (size_t)row*FDIM + p*64;
    bf16* d = A + m*FDIM + p*64;
    for (int t = 0; t < 64; t += 4){
      f32x4 v = *(const f32x4*)(s + t);
      d[t] = (bf16)v[0]; d[t+1] = (bf16)v[1]; d[t+2] = (bf16)v[2]; d[t+3] = (bf16)v[3];
    }
  }
  wave_ds_fence();
  int n = lane & 15, h = lane >> 4;
  for (int nt = 0; nt < 8; ++nt){
    f32x8 acc = gemm_tile(A, FDIM, FDIM, W1T, nt*16, lane);
    float bv = b1[nt*16 + n];
#pragma unroll
    for (int r = 0; r < 8; ++r)
      H[(r + 8*h)*FDIM + nt*16 + n] = (bf16)silu_f(acc[r] + bv);
  }
  wave_ds_fence();
  for (int nt = 0; nt < 8; ++nt){
    f32x8 acc = gemm_tile(H, FDIM, FDIM, W2T, nt*16, lane);
    float bv = b2[nt*16 + n];
#pragma unroll
    for (int r = 0; r < 8; ++r){
      int row = n0 + r + 8*h;
      if (row < N) z1b[(size_t)row*FDIM + nt*16 + n] = (bf16)(acc[r] + bv);
    }
  }
}

// fused edge pipeline: c = concat(z1[i],z1[j])@lc_W+b ; cQ = einsum ; zs = cQ@ls_W+b ;
// dz=zs[:,:F] ; f = silu(silu(zs[:,F:]@mW1+b)@mW2+b) -> f01b,f02b
__global__ void __launch_bounds__(64) k_edge(
    const int* __restrict__ gi, const int* __restrict__ gj,
    const bf16* __restrict__ z1b, const float* __restrict__ Q,
    const bf16* __restrict__ lcWT, const float* __restrict__ lc_b,
    const bf16* __restrict__ lsWT, const float* __restrict__ ls_b,
    const bf16* __restrict__ m1WT, const float* __restrict__ mb1,
    const bf16* __restrict__ m2WT, const float* __restrict__ mb2,
    float* __restrict__ dz, bf16* __restrict__ f01b, bf16* __restrict__ f02b, int E)
{
  __shared__ __align__(32) bf16  sA[2][16*256];
  __shared__ __align__(32) bf16  sS[2][16*256];
  __shared__ __align__(16) float sCq[2][16*FDIM];
  __shared__ __align__(16) float sCt[2][16*16];
  __shared__ __align__(16) float sQ [2][16*CBETA];
  int wave = threadIdx.x >> 5, lane = threadIdx.x & 31;
  bf16* A = sA[wave]; bf16* S = sS[wave];
  float* Cq = sCq[wave]; float* Ct = sCt[wave]; float* Qt = sQ[wave];
  int e0 = (blockIdx.x*2 + wave)*16;

  { // gather concat rows [z1[i] | z1[j]] and the Q tile
    int m = lane >> 1, p = lane & 1;
    int e = e0 + m; if (e > E-1) e = E-1;
    int src = (p == 0) ? gi[e] : gj[e];
    const bf16* s = z1b + (size_t)src*FDIM;
    bf16* d = A + m*256 + p*FDIM;
    for (int t = 0; t < FDIM; t += 8) *(bf16x8*)(d + t) = *(const bf16x8*)(s + t);
    if (p == 0){
      const float* qs = Q + (size_t)e*CBETA;
#pragma unroll
      for (int b = 0; b < CBETA; ++b) Qt[m*CBETA + b] = qs[b];
    }
  }
  wave_ds_fence();
  int n = lane & 15, h = lane >> 4;

  // c tiles (16x16) + Chebyshev contraction -> Cq (16 x 128, f32)
  for (int ft = 0; ft < 8; ++ft){
    for (int ct = 0; ct < 8; ++ct){
      int cb = ft*128 + ct*16;
      f32x8 acc = gemm_tile(A, 256, 256, lcWT, cb, lane);
      float bv = lc_b[cb + n];
#pragma unroll
      for (int r = 0; r < 8; ++r) Ct[(r + 8*h)*16 + n] = acc[r] + bv;
      wave_ds_fence();
      { // col = f*8 + b ; tile covers 2 f values x 8 cheb terms
        int m = lane & 15, fl = lane >> 4;
        float sacc = 0.0f;
#pragma unroll
        for (int b = 0; b < CBETA; ++b) sacc += Ct[m*16 + fl*8 + b] * Qt[m*CBETA + b];
        Cq[m*FDIM + ft*16 + ct*2 + fl] = sacc;
      }
      wave_ds_fence();
    }
  }
  // Cq -> bf16 A-stage (ld = 128)
  for (int t = lane; t < 16*FDIM; t += 32) A[t] = (bf16)Cq[t];
  wave_ds_fence();

  // zs = Cq @ ls_W + b : cols 0..127 -> dz ; cols 128..383 -> f stage (raw)
  for (int nt = 0; nt < 24; ++nt){
    f32x8 acc = gemm_tile(A, FDIM, FDIM, lsWT, nt*16, lane);
    float bv = ls_b[nt*16 + n];
    if (nt < 8){
#pragma unroll
      for (int r = 0; r < 8; ++r){
        int e = e0 + r + 8*h;
        if (e < E) dz[(size_t)e*FDIM + nt*16 + n] = acc[r] + bv;
      }
    } else {
      int cb = (nt - 8)*16;
#pragma unroll
      for (int r = 0; r < 8; ++r) S[(r + 8*h)*256 + cb + n] = (bf16)(acc[r] + bv);
    }
  }
  wave_ds_fence();
  // h = silu(S @ mW1 + b) -> A (ld 256)
  for (int nt = 0; nt < 16; ++nt){
    f32x8 acc = gemm_tile(S, 256, 256, m1WT, nt*16, lane);
    float bv = mb1[nt*16 + n];
#pragma unroll
    for (int r = 0; r < 8; ++r) A[(r + 8*h)*256 + nt*16 + n] = (bf16)silu_f(acc[r] + bv);
  }
  wave_ds_fence();
  // f = silu(A @ mW2 + b) -> f01b | f02b
  for (int nt = 0; nt < 16; ++nt){
    f32x8 acc = gemm_tile(A, 256, 256, m2WT, nt*16, lane);
    float bv = mb2[nt*16 + n];
    int col = nt*16 + n;
#pragma unroll
    for (int r = 0; r < 8; ++r){
      int e = e0 + r + 8*h;
      if (e < E){
        float v = silu_f(acc[r] + bv);
        if (col < FDIM) f01b[(size_t)e*FDIM + col] = (bf16)v;
        else            f02b[(size_t)e*FDIM + col - FDIM] = (bf16)v;
      }
    }
  }
}

// triplets: s01/s02 GEMMs + ms GEMM, silu, atomic segment-sum into Bsum[idx_ji]
__global__ void __launch_bounds__(128) k_triplet(
    const int* __restrict__ idx_ji, const int* __restrict__ idx_ki,
    const float* __restrict__ M01, const float* __restrict__ M02,
    const bf16* __restrict__ f01b, const bf16* __restrict__ f02b,
    const bf16* __restrict__ w01T, const float* __restrict__ b01,
    const bf16* __restrict__ w02T, const float* __restrict__ b02,
    const bf16* __restrict__ msT,  const float* __restrict__ msb,
    float* __restrict__ Bsum, int T)
{
  __shared__ __align__(32) bf16 sG[4][16*FDIM];
  __shared__ __align__(32) bf16 sS[4][16*256];
  int wave = threadIdx.x >> 5, lane = threadIdx.x & 31;
  bf16* G = sG[wave]; bf16* S = sS[wave];
  int t0 = (blockIdx.x*4 + wave)*16;
  int n = lane & 15, h = lane >> 4;

  for (int pass = 0; pass < 2; ++pass){
    const bf16* fb = pass ? f02b : f01b;
    const float* M = pass ? M02  : M01;
    const bf16* wT = pass ? w02T : w01T;
    const float* bb = pass ? b02 : b01;
    { // gather f[idx_ki]*M*f[idx_ji]
      int m = lane >> 1, p = lane & 1;
      int t = t0 + m; if (t > T-1) t = T-1;
      float mm = M[t];
      const bf16* ka = fb + (size_t)idx_ki[t]*FDIM + p*64;
      const bf16* kb = fb + (size_t)idx_ji[t]*FDIM + p*64;
      bf16* d = G + m*FDIM + p*64;
      for (int c = 0; c < 64; ++c) d[c] = (bf16)((float)ka[c] * mm * (float)kb[c]);
    }
    wave_ds_fence();
    for (int nt = 0; nt < 8; ++nt){
      f32x8 acc = gemm_tile(G, FDIM, FDIM, wT, nt*16, lane);
      float bv = bb[nt*16 + n];
#pragma unroll
      for (int r = 0; r < 8; ++r)
        S[(r + 8*h)*256 + pass*FDIM + nt*16 + n] = (bf16)silu_f(acc[r] + bv);
    }
    wave_ds_fence();
  }
  for (int nt = 0; nt < 8; ++nt){
    f32x8 acc = gemm_tile(S, 256, 256, msT, nt*16, lane);
    float bv = msb[nt*16 + n];
#pragma unroll
    for (int r = 0; r < 8; ++r){
      int t = t0 + r + 8*h;
      if (t < T){
        float v = silu_f(acc[r] + bv);
        atomicAdd(&Bsum[(size_t)idx_ji[t]*FDIM + nt*16 + n], v);
      }
    }
  }
}

// B_alpha = Bsum@mo_W+b ; u = silu(dz+B_alpha) ; atomic scatter into nodeacc[i]
__global__ void __launch_bounds__(256) k_edge_out(
    const int* __restrict__ gi, const float* __restrict__ dz, const float* __restrict__ Bsum,
    const bf16* __restrict__ moT, const float* __restrict__ mob,
    float* __restrict__ nodeacc, int E)
{
  __shared__ __align__(32) bf16 sA[8][16*FDIM];
  int wave = threadIdx.x >> 5, lane = threadIdx.x & 31;
  bf16* A = sA[wave];
  int e0 = (blockIdx.x*8 + wave)*16;
  {
    int m = lane >> 1, p = lane & 1;
    int e = e0 + m; if (e > E-1) e = E-1;
    const float* s = Bsum + (size_t)e*FDIM + p*64;
    bf16* d = A + m*FDIM + p*64;
    for (int t = 0; t < 64; t += 4){
      f32x4 v = *(const f32x4*)(s + t);
      d[t] = (bf16)v[0]; d[t+1] = (bf16)v[1]; d[t+2] = (bf16)v[2]; d[t+3] = (bf16)v[3];
    }
  }
  wave_ds_fence();
  int n = lane & 15, h = lane >> 4;
  for (int nt = 0; nt < 8; ++nt){
    f32x8 acc = gemm_tile(A, FDIM, FDIM, moT, nt*16, lane);
    float bv = mob[nt*16 + n];
    int col = nt*16 + n;
#pragma unroll
    for (int r = 0; r < 8; ++r){
      int e = e0 + r + 8*h;
      if (e < E){
        float u = silu_f(dz[(size_t)e*FDIM + col] + acc[r] + bv);
        atomicAdd(&nodeacc[(size_t)gi[e]*FDIM + col], u);
      }
    }
  }
}

// z += nodeacc @ io_W + b
__global__ void __launch_bounds__(256) k_node_out(
    const float* __restrict__ nodeacc, const bf16* __restrict__ ioT,
    const float* __restrict__ iob, float* __restrict__ z, int N)
{
  __shared__ __align__(32) bf16 sA[8][16*FDIM];
  int wave = threadIdx.x >> 5, lane = threadIdx.x & 31;
  bf16* A = sA[wave];
  int n0 = (blockIdx.x*8 + wave)*16;
  {
    int m = lane >> 1, p = lane & 1;
    int row = n0 + m; if (row > N-1) row = N-1;
    const float* s = nodeacc + (size_t)row*FDIM + p*64;
    bf16* d = A + m*FDIM + p*64;
    for (int t = 0; t < 64; t += 4){
      f32x4 v = *(const f32x4*)(s + t);
      d[t] = (bf16)v[0]; d[t+1] = (bf16)v[1]; d[t+2] = (bf16)v[2]; d[t+3] = (bf16)v[3];
    }
  }
  wave_ds_fence();
  int n = lane & 15, h = lane >> 4;
  for (int nt = 0; nt < 8; ++nt){
    f32x8 acc = gemm_tile(A, FDIM, FDIM, ioT, nt*16, lane);
    float bv = iob[nt*16 + n];
#pragma unroll
    for (int r = 0; r < 8; ++r){
      int row = n0 + r + 8*h;
      if (row < N) z[(size_t)row*FDIM + nt*16 + n] += acc[r] + bv;
    }
  }
}

// out = silu(z@en_W1+b1)@en_W2+b2
__global__ void __launch_bounds__(128) k_energy(
    const float* __restrict__ z, const bf16* __restrict__ e1T, const float* __restrict__ eb1,
    const float* __restrict__ eW2, const float* __restrict__ eb2,
    float* __restrict__ out, int N)
{
  __shared__ __align__(32) bf16  sA[4][16*FDIM];
  __shared__ __align__(16) float sH[4][16*FDIM];
  int wave = threadIdx.x >> 5, lane = threadIdx.x & 31;
  bf16* A = sA[wave]; float* H = sH[wave];
  int n0 = (blockIdx.x*4 + wave)*16;
  {
    int m = lane >> 1, p = lane & 1;
    int row = n0 + m; if (row > N-1) row = N-1;
    const float* s = z + (size_t)row*FDIM + p*64;
    bf16* d = A + m*FDIM + p*64;
    for (int t = 0; t < 64; t += 4){
      f32x4 v = *(const f32x4*)(s + t);
      d[t] = (bf16)v[0]; d[t+1] = (bf16)v[1]; d[t+2] = (bf16)v[2]; d[t+3] = (bf16)v[3];
    }
  }
  wave_ds_fence();
  int n = lane & 15, h = lane >> 4;
  for (int nt = 0; nt < 8; ++nt){
    f32x8 acc = gemm_tile(A, FDIM, FDIM, e1T, nt*16, lane);
    float bv = eb1[nt*16 + n];
#pragma unroll
    for (int r = 0; r < 8; ++r) H[(r + 8*h)*FDIM + nt*16 + n] = silu_f(acc[r] + bv);
  }
  wave_ds_fence();
  if (lane < 16){
    int row = n0 + lane;
    if (row < N){
      float s = eb2[0];
      for (int k = 0; k < FDIM; ++k) s += H[lane*FDIM + k]*eW2[k];
      out[row] = s;
    }
  }
}

// ---------------- host ----------------

extern "C" void kernel_launch(void* const* d_in, const int* in_sizes, int n_in,
                              void* d_out, int out_size, void* d_ws, size_t ws_size,
                              hipStream_t stream)
{
  const int*   an     = (const int*)d_in[0];
  const int*   gi     = (const int*)d_in[1];
  const int*   gj     = (const int*)d_in[2];
  const float* dist   = (const float*)d_in[3];
  const float* M01    = (const float*)d_in[4];
  const float* M02    = (const float*)d_in[5];
  const int*   idx_ji = (const int*)d_in[6];
  const int*   idx_ki = (const int*)d_in[7];
  const float* emb    = (const float*)d_in[8];
  const float* lz_W1  = (const float*)d_in[9];
  const float* lz_b1  = (const float*)d_in[10];
  const float* lz_W2  = (const float*)d_in[11];
  const float* lz_b2  = (const float*)d_in[12];
  const float* lc_W   = (const float*)d_in[13];
  const float* lc_b   = (const float*)d_in[14];
  const float* ls_W   = (const float*)d_in[15];
  const float* ls_b   = (const float*)d_in[16];
  const float* m_W1   = (const float*)d_in[17];
  const float* m_b1   = (const float*)d_in[18];
  const float* m_W2   = (const float*)d_in[19];
  const float* m_b2   = (const float*)d_in[20];
  const float* m01_W  = (const float*)d_in[21];
  const float* m01_b  = (const float*)d_in[22];
  const float* m02_W  = (const float*)d_in[23];
  const float* m02_b  = (const float*)d_in[24];
  const float* ms_W   = (const float*)d_in[25];
  const float* ms_b   = (const float*)d_in[26];
  const float* mo_W   = (const float*)d_in[27];
  const float* mo_b   = (const float*)d_in[28];
  const float* io_W   = (const float*)d_in[29];
  const float* io_b   = (const float*)d_in[30];
  const float* en_W1  = (const float*)d_in[31];
  const float* en_b1  = (const float*)d_in[32];
  const float* en_W2  = (const float*)d_in[33];
  const float* en_b2  = (const float*)d_in[34];
  (void)in_sizes; (void)n_in; (void)out_size; (void)ws_size;

  char* base = (char*)d_ws;
  size_t off = 0;
  auto alloc = [&](size_t bytes)->void*{
    off = (off + 255) & ~(size_t)255;
    void* p = base + off; off += bytes; return p;
  };
  float* Q    = (float*)alloc((size_t)N_EDGES*CBETA*4);
  float* z    = (float*)alloc((size_t)N_NODES*FDIM*4);
  bf16*  z1b  = (bf16*) alloc((size_t)N_NODES*FDIM*2);
  float* dz   = (float*)alloc((size_t)N_EDGES*FDIM*4);
  bf16*  f01b = (bf16*) alloc((size_t)N_EDGES*FDIM*2);
  bf16*  f02b = (bf16*) alloc((size_t)N_EDGES*FDIM*2);
  float* Bsum = (float*)alloc((size_t)N_EDGES*FDIM*4);
  float* nacc = (float*)alloc((size_t)N_NODES*FDIM*4);

  const size_t SZ_FF  = (size_t)FDIM*FDIM;        // 16384
  const size_t SZ_LC  = (size_t)2*FDIM*FDIM*CBETA;// 262144
  const size_t SZ_LS  = (size_t)FDIM*3*FDIM;      // 49152
  const size_t SZ_MM  = (size_t)2*FDIM*2*FDIM;    // 65536
  const size_t SZ_MS  = (size_t)2*FDIM*FDIM;      // 32768
  bf16* t_lzW1 = (bf16*)alloc(NLAYERS*SZ_FF*2);
  bf16* t_lzW2 = (bf16*)alloc(NLAYERS*SZ_FF*2);
  bf16* t_lcW  = (bf16*)alloc(NLAYERS*SZ_LC*2);
  bf16* t_lsW  = (bf16*)alloc(NLAYERS*SZ_LS*2);
  bf16* t_mW1  = (bf16*)alloc(NLAYERS*SZ_MM*2);
  bf16* t_mW2  = (bf16*)alloc(NLAYERS*SZ_MM*2);
  bf16* t_m01  = (bf16*)alloc(NLAYERS*SZ_FF*2);
  bf16* t_m02  = (bf16*)alloc(NLAYERS*SZ_FF*2);
  bf16* t_ms   = (bf16*)alloc(NLAYERS*SZ_MS*2);
  bf16* t_mo   = (bf16*)alloc(NLAYERS*SZ_FF*2);
  bf16* t_io   = (bf16*)alloc(NLAYERS*SZ_FF*2);
  bf16* t_en1  = (bf16*)alloc(SZ_FF*2);

  auto cvt = [&](const float* W, bf16* WT, int K, int Nout){
    int tot = K*Nout;
    k_cvtT<<<(tot + 255)/256, 256, 0, stream>>>(W, WT, K, Nout);
  };
  for (int l = 0; l < NLAYERS; ++l){
    cvt(lz_W1 + l*SZ_FF, t_lzW1 + l*SZ_FF, FDIM, FDIM);
    cvt(lz_W2 + l*SZ_FF, t_lzW2 + l*SZ_FF, FDIM, FDIM);
    cvt(lc_W  + l*SZ_LC, t_lcW  + l*SZ_LC, 2*FDIM, FDIM*CBETA);
    cvt(ls_W  + l*SZ_LS, t_lsW  + l*SZ_LS, FDIM, 3*FDIM);
    cvt(m_W1  + l*SZ_MM, t_mW1  + l*SZ_MM, 2*FDIM, 2*FDIM);
    cvt(m_W2  + l*SZ_MM, t_mW2  + l*SZ_MM, 2*FDIM, 2*FDIM);
    cvt(m01_W + l*SZ_FF, t_m01  + l*SZ_FF, FDIM, FDIM);
    cvt(m02_W + l*SZ_FF, t_m02  + l*SZ_FF, FDIM, FDIM);
    cvt(ms_W  + l*SZ_MS, t_ms   + l*SZ_MS, 2*FDIM, FDIM);
    cvt(mo_W  + l*SZ_FF, t_mo   + l*SZ_FF, FDIM, FDIM);
    cvt(io_W  + l*SZ_FF, t_io   + l*SZ_FF, FDIM, FDIM);
  }
  cvt(en_W1, t_en1, FDIM, FDIM);

  k_cheb <<<(N_EDGES + 255)/256, 256, 0, stream>>>(dist, Q, N_EDGES);
  k_embed<<<(N_NODES*FDIM + 255)/256, 256, 0, stream>>>(an, emb, z, N_NODES);

  const int nodeTiles = (N_NODES + 15)/16;   // 625
  const int edgeTiles = (N_EDGES + 15)/16;   // 7500
  const int tripTiles = (N_TRIP  + 15)/16;   // 37500

  for (int l = 0; l < NLAYERS; ++l){
    k_zero<<<2048, 256, 0, stream>>>(Bsum, (size_t)N_EDGES*FDIM);
    k_zero<<<2048, 256, 0, stream>>>(nacc, (size_t)N_NODES*FDIM);

    k_node_mlp<<<(nodeTiles + 3)/4, 128, 0, stream>>>(
        z, t_lzW1 + l*SZ_FF, lz_b1 + l*FDIM, t_lzW2 + l*SZ_FF, lz_b2 + l*FDIM, z1b, N_NODES);

    k_edge<<<(edgeTiles + 1)/2, 64, 0, stream>>>(
        gi, gj, z1b, Q,
        t_lcW + l*SZ_LC, lc_b + l*FDIM*CBETA,
        t_lsW + l*SZ_LS, ls_b + l*3*FDIM,
        t_mW1 + l*SZ_MM, m_b1 + l*2*FDIM,
        t_mW2 + l*SZ_MM, m_b2 + l*2*FDIM,
        dz, f01b, f02b, N_EDGES);

    k_triplet<<<(tripTiles + 3)/4, 128, 0, stream>>>(
        idx_ji, idx_ki, M01, M02, f01b, f02b,
        t_m01 + l*SZ_FF, m01_b + l*FDIM,
        t_m02 + l*SZ_FF, m02_b + l*FDIM,
        t_ms  + l*SZ_MS, ms_b  + l*FDIM,
        Bsum, N_TRIP);

    k_edge_out<<<(edgeTiles + 7)/8, 256, 0, stream>>>(
        gi, dz, Bsum, t_mo + l*SZ_FF, mo_b + l*FDIM, nacc, N_EDGES);

    k_node_out<<<(nodeTiles + 7)/8, 256, 0, stream>>>(
        nacc, t_io + l*SZ_FF, io_b + l*FDIM, z, N_NODES);
  }

  k_energy<<<(nodeTiles + 3)/4, 128, 0, stream>>>(
      z, t_en1, en_b1, en_W2, en_b2, (float*)d_out, N_NODES);
}
